// adapt_gcn_47931835023749
// MI455X (gfx1250) — matrix operand order repaired and checked by reading, not compiled, not run on hardware
//
#include <hip/hip_runtime.h>
#include <hip/hip_bf16.h>
#include <math.h>

typedef __attribute__((ext_vector_type(16))) __bf16          v16bf;
typedef __attribute__((ext_vector_type(8)))  float           v8f;
typedef __attribute__((ext_vector_type(8)))  unsigned short  us8;
typedef __attribute__((ext_vector_type(4)))  unsigned short  us4;
typedef __attribute__((ext_vector_type(4)))  float           f4;

#define NB   16
#define CCH  256
#define TT   256
#define VV   25
#define VPP  20
#define SS   3
#define ICC  64
#define NCOLS (TT * VPP)   // 5120

__device__ __forceinline__ unsigned short f2bf(float f) {
  unsigned int u = __float_as_uint(f);
  unsigned int r = u + 0x7FFFu + ((u >> 16) & 1u);   // round-to-nearest-even
  return (unsigned short)(r >> 16);
}
__device__ __forceinline__ float bf2f(unsigned short h) {
  return __uint_as_float(((unsigned int)h) << 16);
}

// ---------------------------------------------------------------------------
// K1: pack weights to bf16 layouts, fold biases, A = A_fix + PA
// ---------------------------------------------------------------------------
__global__ __launch_bounds__(256)
void k_prep(const float* __restrict__ PA, const float* __restrict__ A_fix,
            const float* __restrict__ wa, const float* __restrict__ ba,
            const float* __restrict__ wb, const float* __restrict__ bbp,
            const float* __restrict__ wd, const float* __restrict__ bd,
            unsigned short* __restrict__ wab, unsigned short* __restrict__ wdall,
            float* __restrict__ bab, float* __restrict__ bdsum,
            float* __restrict__ Acomb) {
  int i = blockIdx.x * 256 + threadIdx.x;
  if (i < SS * 128 * CCH) {          // [wa;wb] per subset: 128 x 256 rows
    int s = i / (128 * CCH); int r = (i / CCH) % 128; int c = i % CCH;
    float v = (r < ICC) ? wa[(s * ICC + r) * CCH + c]
                        : wb[(s * ICC + (r - ICC)) * CCH + c];
    wab[i] = f2bf(v);
  }
  if (i < CCH * SS * CCH) {          // wd_all[o][s*256+c]
    int o = i / (SS * CCH); int k = i % (SS * CCH);
    int s = k / CCH; int c = k % CCH;
    wdall[i] = f2bf(wd[((s * CCH) + o) * CCH + c]);
  }
  if (i < SS * 128) {
    int s = i / 128, r = i % 128;
    bab[i] = (r < ICC) ? ba[s * ICC + r] : bbp[s * ICC + r - ICC];
  }
  if (i < CCH) bdsum[i] = bd[i] + bd[CCH + i] + bd[2 * CCH + i];
  if (i < SS * VPP * VPP) Acomb[i] = A_fix[i] + PA[i];
}

// ---------------------------------------------------------------------------
// K2: xp = x @ w_e + b_e   (V=25 -> VP=20 node remap), fp32 out
// ---------------------------------------------------------------------------
__global__ __launch_bounds__(256)
void k_remap(const float* __restrict__ x, const float* __restrict__ w_e,
             const float* __restrict__ b_e, float* __restrict__ xp) {
  __shared__ float wl[VV * VPP + VPP];
  for (int i = threadIdx.x; i < VV * VPP + VPP; i += 256)
    wl[i] = (i < VV * VPP) ? w_e[i] : b_e[i - VV * VPP];
  __syncthreads();
  long long gid = (long long)blockIdx.x * 256 + threadIdx.x;  // (n,c,t) row
  const float* xr = x + gid * VV;
  float xv[VV];
#pragma unroll
  for (int v = 0; v < VV; ++v) xv[v] = xr[v];
  float outv[VPP];
#pragma unroll
  for (int u = 0; u < VPP; ++u) outv[u] = wl[VV * VPP + u];
#pragma unroll
  for (int v = 0; v < VV; ++v)
#pragma unroll
    for (int u = 0; u < VPP; ++u)
      outv[u] = fmaf(xv[v], wl[v * VPP + u], outv[u]);
  float* op = xp + gid * VPP;
#pragma unroll
  for (int u = 0; u < VPP; u += 4) *(f4*)(op + u) = *(const f4*)(outv + u);
}

// ---------------------------------------------------------------------------
// WMMA bf16 GEMM: Out[M x 5120] = A[M x K](bf16) * B[K x 5120] + bias[row]
//   128x128 block tile, BK=32, 8 waves (2x4), 4x2 wmma tiles per wave.
//   B_IS_F32: convert B tile to bf16 while staging to LDS.
//   OUT_BF16: emit bf16 (fa/fb) else fp32 (y).
// ---------------------------------------------------------------------------
template <bool B_IS_F32, bool OUT_BF16>
__global__ __launch_bounds__(256)
void gemm128_wmma(const unsigned short* __restrict__ A,
                  const void* __restrict__ Bv, void* __restrict__ Outv,
                  const float* __restrict__ bias, int K, int a_mod, int b_div,
                  long long a_stride, long long b_stride,
                  long long out_stride, long long bias_stride) {
  __shared__ unsigned short Alds[128][32];   // [row][k]
  __shared__ unsigned short Blds[128][32];   // [col][k]  (transposed tile)

  const int tid = threadIdx.x;
  const int batch = blockIdx.z;
  const int mbase = blockIdx.y * 128;
  const int nbase = blockIdx.x * 128;
  const unsigned short* Ab = A + (long long)(batch % a_mod) * a_stride;
  const float* bias_b = bias + (long long)(batch % a_mod) * bias_stride;

  const int wid = tid >> 5, lane = tid & 31;
  const int wm = wid >> 2, wn = wid & 3;

  v8f acc[4][2];
#pragma unroll
  for (int i = 0; i < 4; ++i)
#pragma unroll
    for (int j = 0; j < 2; ++j) acc[i][j] = (v8f)0.0f;

  const int a_row = tid >> 1, a_koff = (tid & 1) * 16;
  const int b_krow = tid >> 3, b_coff = (tid & 7) * 16;
  const int ksteps = K >> 5;

  for (int ks = 0; ks < ksteps; ++ks) {
    const int kb = ks << 5;
    // ---- stage A tile (bf16, row-major) ----
    {
      const unsigned short* src = Ab + (long long)(mbase + a_row) * K + kb + a_koff;
      us8 lo = *(const us8*)(src);
      us8 hi = *(const us8*)(src + 8);
      *(us8*)&Alds[a_row][a_koff]     = lo;
      *(us8*)&Alds[a_row][a_koff + 8] = hi;
    }
    // ---- stage B tile transposed into LDS ----
    if (B_IS_F32) {
      const float* Bf = (const float*)Bv + (long long)(batch / b_div) * b_stride;
      const float* src = Bf + (long long)(kb + b_krow) * NCOLS + nbase + b_coff;
      if (ks + 1 < ksteps) __builtin_prefetch((const void*)(src + 32LL * NCOLS), 0, 1);
#pragma unroll
      for (int j = 0; j < 16; ++j) Blds[b_coff + j][b_krow] = f2bf(src[j]);
    } else {
      const unsigned short* Bh =
          (const unsigned short*)Bv + (long long)(batch / b_div) * b_stride;
      const unsigned short* src = Bh + (long long)(kb + b_krow) * NCOLS + nbase + b_coff;
      if (ks + 1 < ksteps) __builtin_prefetch((const void*)(src + 32LL * NCOLS), 0, 1);
      us8 lo = *(const us8*)src;
      us8 hi = *(const us8*)(src + 8);
#pragma unroll
      for (int j = 0; j < 8; ++j) {
        Blds[b_coff + j][b_krow]     = lo[j];
        Blds[b_coff + 8 + j][b_krow] = hi[j];
      }
    }
    __syncthreads();

    // ---- fragments, contiguous k order (k = 16*half + s), same for A and B ----
    union { v16bf v; us8 h[2]; } bf[2];
#pragma unroll
    for (int j = 0; j < 2; ++j) {
      const int col = wn * 32 + j * 16 + (lane & 15);
      const int kst = (lane >> 4) * 16;            // K 0-15 | 16-31 by lane half
      bf[j].h[0] = *(const us8*)&Blds[col][kst];
      bf[j].h[1] = *(const us8*)&Blds[col][kst + 8];
    }
#pragma unroll
    for (int i = 0; i < 4; ++i) {
      union { v16bf v; us8 h[2]; } af;
      const int row = wm * 64 + i * 16 + (lane & 15);
      const int kst = (lane >> 4) * 16;            // K 0-15 | 16-31 by lane half
      af.h[0] = *(const us8*)&Alds[row][kst];
      af.h[1] = *(const us8*)&Alds[row][kst + 8];
#pragma unroll
      for (int j = 0; j < 2; ++j)
        acc[i][j] = __builtin_amdgcn_wmma_f32_16x16x32_bf16(
            false, af.v, false, bf[j].v, (short)0, acc[i][j], false, false);
    }
    __syncthreads();
  }

  // ---- epilogue: D layout vgpr r -> row r (+8 for upper lane half) ----
  const int ncol = lane & 15;
  const int rsel = (lane >> 4) * 8;
#pragma unroll
  for (int i = 0; i < 4; ++i)
#pragma unroll
    for (int j = 0; j < 2; ++j)
#pragma unroll
      for (int r = 0; r < 8; ++r) {
        const int row = wm * 64 + i * 16 + rsel + r;
        const int col = wn * 32 + j * 16 + ncol;
        const long long o = (long long)batch * out_stride +
                            (long long)(mbase + row) * NCOLS + (nbase + col);
        const float v = acc[i][j][r] + bias_b[mbase + row];
        if (OUT_BF16) ((unsigned short*)Outv)[o] = f2bf(v);
        else          ((float*)Outv)[o] = v;
      }
}

// ---------------------------------------------------------------------------
// K4: scores[n,s,v,u] = (1/(IC*T)) * sum_{i,t} fa[..i,t,v] * fb[..i,t,u]
// one block per output element; K=16384 split across 256 threads
// ---------------------------------------------------------------------------
__global__ __launch_bounds__(256)
void k_scores(const unsigned short* __restrict__ fafb, float* __restrict__ scores) {
  __shared__ float red[256];
  const int b = blockIdx.x;
  const int u = b % VPP, v = (b / VPP) % VPP, ns = b / (VPP * VPP);
  const unsigned short* base = fafb + (long long)ns * 128 * NCOLS;
  const int t = threadIdx.x;                      // one t per thread per i
  float sum = 0.f;
  for (int i = 0; i < ICC; ++i) {
    float a = bf2f(base[(long long)i * NCOLS + t * VPP + v]);
    float c = bf2f(base[(long long)(ICC + i) * NCOLS + t * VPP + u]);
    sum = fmaf(a, c, sum);
  }
  red[t] = sum; __syncthreads();
  for (int off = 128; off > 0; off >>= 1) {
    if (t < off) red[t] += red[t + off];
    __syncthreads();
  }
  if (t == 0) scores[b] = red[0] * (1.0f / (ICC * TT));
}

// ---------------------------------------------------------------------------
// K5: att = softmax over v (dim -2) + (A_fix + PA)
// ---------------------------------------------------------------------------
__global__ __launch_bounds__(256)
void k_softmax(const float* __restrict__ scores, const float* __restrict__ Acomb,
               float* __restrict__ att) {
  int gid = blockIdx.x * 256 + threadIdx.x;
  if (gid >= NB * SS * VPP) return;
  const int u = gid % VPP, s = (gid / VPP) % SS, n = gid / (VPP * SS);
  const float* sc = scores + ((long long)(n * SS + s) * VPP) * VPP + u;
  float vals[VPP], mx = -1e30f;
#pragma unroll
  for (int v = 0; v < VPP; ++v) { vals[v] = sc[v * VPP]; mx = fmaxf(mx, vals[v]); }
  float sum = 0.f;
#pragma unroll
  for (int v = 0; v < VPP; ++v) { vals[v] = __expf(vals[v] - mx); sum += vals[v]; }
  const float inv = 1.f / sum;
  float* ao = att + ((long long)(n * SS + s) * VPP) * VPP + u;
#pragma unroll
  for (int v = 0; v < VPP; ++v)
    ao[v * VPP] = vals[v] * inv + Acomb[(s * VPP + v) * VPP + u];
}

// ---------------------------------------------------------------------------
// K6: z[n,s,c,t,u] = sum_v xp[n,c,t,v] * att[n,s,v,u]   (bf16 out)
// ---------------------------------------------------------------------------
__global__ __launch_bounds__(256)
void k_zmix(const float* __restrict__ xp, const float* __restrict__ att,
            unsigned short* __restrict__ z) {
  __shared__ float al[SS * VPP * VPP];
  const int n = blockIdx.y;
  for (int i = threadIdx.x; i < SS * VPP * VPP; i += 256)
    al[i] = att[(long long)n * SS * VPP * VPP + i];
  __syncthreads();
  const int row = blockIdx.x * 256 + threadIdx.x;  // c*T + t
  const int c = row >> 8, t = row & 255;
  const float* xr = xp + ((long long)(n * CCH + c) * TT + t) * VPP;
  float xv[VPP];
#pragma unroll
  for (int v = 0; v < VPP; ++v) xv[v] = xr[v];
#pragma unroll
  for (int s = 0; s < SS; ++s) {
    unsigned short ov[VPP];
#pragma unroll
    for (int u = 0; u < VPP; ++u) {
      float a = 0.f;
#pragma unroll
      for (int v = 0; v < VPP; ++v) a = fmaf(xv[v], al[(s * VPP + v) * VPP + u], a);
      ov[u] = f2bf(a);
    }
    unsigned short* zp = z + ((long long)(n * SS + s) * CCH + c) * NCOLS + t * VPP;
#pragma unroll
    for (int u = 0; u < VPP; u += 4) *(us4*)(zp + u) = *(const us4*)(ov + u);
  }
}

// ---------------------------------------------------------------------------
// K8: per-channel batch statistics of y -> mean, rstd
// ---------------------------------------------------------------------------
__global__ __launch_bounds__(256)
void k_stats(const float* __restrict__ y, float* __restrict__ stats) {
  __shared__ float rs[256], rq[256];
  const int o = blockIdx.x;
  float s = 0.f, q = 0.f;
  for (int n = 0; n < NB; ++n) {
    const float* yr = y + (long long)(n * CCH + o) * NCOLS;
    for (int col = threadIdx.x; col < NCOLS; col += 256) {
      float v = yr[col]; s += v; q = fmaf(v, v, q);
    }
  }
  rs[threadIdx.x] = s; rq[threadIdx.x] = q; __syncthreads();
  for (int off = 128; off > 0; off >>= 1) {
    if (threadIdx.x < off) {
      rs[threadIdx.x] += rs[threadIdx.x + off];
      rq[threadIdx.x] += rq[threadIdx.x + off];
    }
    __syncthreads();
  }
  if (threadIdx.x == 0) {
    const float cnt = (float)(NB * NCOLS);
    const float mean = rs[0] / cnt;
    const float var = rq[0] / cnt - mean * mean;
    stats[o] = mean;
    stats[CCH + o] = rsqrtf(var + 1e-5f);
  }
}

// ---------------------------------------------------------------------------
// K9: out = relu( (y - mean)*rstd*gamma + beta + xp )
// ---------------------------------------------------------------------------
__global__ __launch_bounds__(256)
void k_final(const float* __restrict__ y, const float* __restrict__ xp,
             const float* __restrict__ stats, const float* __restrict__ gamma,
             const float* __restrict__ beta, float* __restrict__ out) {
  const long long idx = (long long)blockIdx.x * 256 + threadIdx.x;
  const int c = (int)((idx / NCOLS) % CCH);
  const float v =
      (y[idx] - stats[c]) * stats[CCH + c] * gamma[c] + beta[c] + xp[idx];
  out[idx] = fmaxf(v, 0.f);
}

// ---------------------------------------------------------------------------
extern "C" void kernel_launch(void* const* d_in, const int* in_sizes, int n_in,
                              void* d_out, int out_size, void* d_ws, size_t ws_size,
                              hipStream_t stream) {
  (void)in_sizes; (void)n_in; (void)out_size; (void)ws_size;
  const float* x     = (const float*)d_in[0];
  const float* PA    = (const float*)d_in[1];
  const float* A_fix = (const float*)d_in[2];
  const float* w_e   = (const float*)d_in[3];
  const float* b_e   = (const float*)d_in[4];
  const float* wa    = (const float*)d_in[5];
  const float* ba    = (const float*)d_in[6];
  const float* wb    = (const float*)d_in[7];
  const float* bb    = (const float*)d_in[8];
  const float* wd    = (const float*)d_in[9];
  const float* bd    = (const float*)d_in[10];
  const float* gamma = (const float*)d_in[11];
  const float* beta  = (const float*)d_in[12];
  float* out = (float*)d_out;

  char* ws = (char*)d_ws;
  const size_t XP_BYTES  = (size_t)NB * CCH * TT * VPP * 4;   // 83,886,080
  const size_t BIG_BYTES = (size_t)NB * SS * CCH * NCOLS * 2; // 125,829,120
  size_t off = 0;
  auto take = [&](size_t bytes) { size_t o = off; off += (bytes + 255) & ~(size_t)255; return o; };
  const size_t o_xp  = take(XP_BYTES);                 // fp32 xp
  const size_t o_y   = take(XP_BYTES);                 // fp32 y
  const size_t o_big = take(BIG_BYTES);                // bf16 fafb, later bf16 z
  const size_t o_wab = take((size_t)SS * 128 * CCH * 2);
  const size_t o_wd  = take((size_t)CCH * SS * CCH * 2);
  const size_t o_bab = take((size_t)SS * 128 * 4);
  const size_t o_bds = take((size_t)CCH * 4);
  const size_t o_A   = take((size_t)SS * VPP * VPP * 4);
  const size_t o_sc  = take((size_t)NB * SS * VPP * VPP * 4);
  const size_t o_att = take((size_t)NB * SS * VPP * VPP * 4);
  const size_t o_st  = take((size_t)2 * CCH * 4);

  float*          xp_f  = (float*)(ws + o_xp);
  float*          y_f   = (float*)(ws + o_y);
  unsigned short* big_h = (unsigned short*)(ws + o_big);
  unsigned short* wab_h = (unsigned short*)(ws + o_wab);
  unsigned short* wd_h  = (unsigned short*)(ws + o_wd);
  float*          bab_f = (float*)(ws + o_bab);
  float*          bds_f = (float*)(ws + o_bds);
  float*          A_f   = (float*)(ws + o_A);
  float*          sc_f  = (float*)(ws + o_sc);
  float*          att_f = (float*)(ws + o_att);
  float*          st_f  = (float*)(ws + o_st);

  // K1: weight packing
  k_prep<<<768, 256, 0, stream>>>(PA, A_fix, wa, ba, wb, bb, wd, bd,
                                  wab_h, wd_h, bab_f, bds_f, A_f);
  // K2: node remap -> xp
  k_remap<<<(NB * CCH * TT) / 256, 256, 0, stream>>>(x, w_e, b_e, xp_f);
  // K3: fa|fb = [wa;wb] @ xp  (WMMA, B fp32 -> bf16, bf16 out), batch = n*S+s
  gemm128_wmma<true, true><<<dim3(NCOLS / 128, 1, NB * SS), 256, 0, stream>>>(
      wab_h, (const void*)xp_f, (void*)big_h, bab_f,
      /*K=*/CCH, /*a_mod=*/SS, /*b_div=*/SS,
      /*a_stride=*/(long long)128 * CCH,
      /*b_stride=*/(long long)CCH * NCOLS,
      /*out_stride=*/(long long)128 * NCOLS,
      /*bias_stride=*/128);
  // K4: attention scores
  k_scores<<<NB * SS * VPP * VPP, 256, 0, stream>>>(big_h, sc_f);
  // K5: softmax(dim=-2) + A
  k_softmax<<<(NB * SS * VPP + 255) / 256, 256, 0, stream>>>(sc_f, A_f, att_f);
  // K6: z = xp @ att  (bf16 out; overwrites fafb region after it is consumed)
  k_zmix<<<dim3((CCH * TT) / 256, NB), 256, 0, stream>>>(xp_f, att_f, big_h);
  // K7: y = wd_all @ z + bd_sum  (WMMA, bf16 in, fp32 out), batch = n
  gemm128_wmma<false, false><<<dim3(NCOLS / 128, CCH / 128, NB), 256, 0, stream>>>(
      wd_h, (const void*)big_h, (void*)y_f, bds_f,
      /*K=*/SS * CCH, /*a_mod=*/1, /*b_div=*/1,
      /*a_stride=*/0,
      /*b_stride=*/(long long)SS * CCH * NCOLS,
      /*out_stride=*/(long long)CCH * NCOLS,
      /*bias_stride=*/0);
  // K8: batch-norm statistics
  k_stats<<<CCH, 256, 0, stream>>>(y_f, st_f);
  // K9: normalize + residual + relu
  k_final<<<(NB * CCH * TT * VPP) / 256, 256, 0, stream>>>(
      y_f, xp_f, st_f, gamma, beta, out);
}
